// Decoder_49349174231051
// MI455X (gfx1250) — matrix operand order repaired and checked by reading, not compiled, not run on hardware
//
#include <hip/hip_runtime.h>
#include <hip/hip_bf16.h>
#include <math.h>

// Problem sizes (fixed by the reference)
#define Bsz 256
#define Tsz 128
#define Msz 64
#define Psz 64

typedef __attribute__((ext_vector_type(16))) __bf16 v16bf;
typedef __attribute__((ext_vector_type(8)))  float  v8f;

// ---------------------------------------------------------------------------
// WMMA GEMM helper: one 16x16 f32 tile of  C = A(16xK) * B(Kx16) + bias,
// K = 64, bf16 inputs, f32 accumulate, via v_wmma_f32_16x16x32_bf16 (x2).
// A row-major (lda = K), B row-major (ldb = N_total).
// VGPR layouts per CDNA5 ISA 7.12.2:
//   A 16x32 (16-bit): lane row = lane&15; elem e -> k = (e&7) + (e>>3)*16 + (lane>>4)*8
//   B 32x16 (16-bit): elem e -> k = (e&7) + (e>>3)*16 + (lane>>4)*8, col = lane&15
//   C/D 8xf32:        elem i -> row = i + (lane>>4)*8, col = lane&15
// ---------------------------------------------------------------------------
__device__ __forceinline__ void wmma_tile_k64(
    const float* __restrict__ A, int lda, int rowBase,
    const float* __restrict__ Bm, int ldb, int colBase,
    const float* __restrict__ bias,
    float* __restrict__ C, int ldc)
{
  const int lane = threadIdx.x & 31;
  const int hl   = lane >> 4;     // lane half (0/1)
  const int l15  = lane & 15;

  v8f acc = {};
  #pragma unroll
  for (int kb = 0; kb < 64; kb += 32) {
    v16bf a, b;
    const float* arow = A + (size_t)(rowBase + l15) * lda + kb + hl * 8;
    #pragma unroll
    for (int e = 0; e < 8; ++e) a[e]     = (__bf16)arow[e];        // k = kb+hl*8+e
    #pragma unroll
    for (int e = 0; e < 8; ++e) a[8 + e] = (__bf16)arow[16 + e];   // k = kb+16+hl*8+e
    const float* bcol = Bm + (size_t)(kb + hl * 8) * ldb + colBase + l15;
    #pragma unroll
    for (int e = 0; e < 16; ++e) b[e] = (__bf16)bcol[(size_t)(e + (e & 8)) * ldb];
    acc = __builtin_amdgcn_wmma_f32_16x16x32_bf16(
        false, a, false, b, (short)0, acc, false, false);
  }
  const float bs = bias ? bias[colBase + l15] : 0.0f;
  #pragma unroll
  for (int i = 0; i < 8; ++i) {
    const int r = rowBase + hl * 8 + i;
    C[(size_t)r * ldc + colBase + l15] = acc[i] + bs;
  }
}

// Kernel 1: Ue = encoder_h(2d: 32768x64) @ Ud(64x64) + bUd   (one-shot, WMMA)
__global__ __launch_bounds__(256)
void ue_gemm_kernel(const float* __restrict__ enc, const float* __restrict__ Ud,
                    const float* __restrict__ bUd, float* __restrict__ Ue)
{
  const int wave = blockIdx.x * (blockDim.x >> 5) + (threadIdx.x >> 5);
  const int tn = wave & 3;            // 4 col tiles (64 cols)
  const int tm = wave >> 2;           // 2048 row tiles (32768 rows)
  wmma_tile_k64(enc, 64, tm * 16, Ud, 64, tn * 16, bUd, Ue, 64);
}

// Kernel 2: H0U = h0(256x64) @ Uk(64x256) + bk   (one-shot, WMMA)
__global__ __launch_bounds__(256)
void h0u_gemm_kernel(const float* __restrict__ h0, const float* __restrict__ Uk,
                     const float* __restrict__ bk, float* __restrict__ H0U)
{
  const int wave = blockIdx.x * (blockDim.x >> 5) + (threadIdx.x >> 5);
  const int tn = wave & 15;           // 16 col tiles (256 cols)
  const int tm = wave >> 4;           // 16 row tiles (256 rows)
  wmma_tile_k64(h0, 64, tm * 16, Uk, 256, tn * 16, bk, H0U, 256);
}

// ---------------------------------------------------------------------------
// Kernel 3: persistent per-batch-row decoder scan (127 steps inside the block)
// ---------------------------------------------------------------------------
__device__ __forceinline__ float wave_sum(float v) {
  #pragma unroll
  for (int off = 16; off > 0; off >>= 1) v += __shfl_down(v, off, 32);
  return v;
}
__device__ __forceinline__ float wave_max(float v) {
  #pragma unroll
  for (int off = 16; off > 0; off >>= 1) v = fmaxf(v, __shfl_down(v, off, 32));
  return v;
}
__device__ __forceinline__ float sigf(float x) { return 1.0f / (1.0f + __expf(-x)); }

__global__ __launch_bounds__(128)
void decoder_scan_kernel(const float* __restrict__ data,      // (B,127)
                         const float* __restrict__ enc,       // (B,T,M)
                         const float* __restrict__ s0,        // (B,P)
                         const float* __restrict__ Wdense,    // (65)
                         const float* __restrict__ bdense_p,  // (1)
                         const float* __restrict__ Wk,        // (256)
                         const float* __restrict__ Wd,        // (128,64)
                         const float* __restrict__ bWd,       // (64)
                         const float* __restrict__ vd,        // (64)
                         const float* __restrict__ bvd_p,     // (1)
                         const float* __restrict__ Wvb,       // (128)
                         const float* __restrict__ bvb_p,     // (1)
                         const float* __restrict__ Wb,        // (64)
                         const float* __restrict__ bWb,       // (64)
                         const float* __restrict__ Ue,        // (B,T,M) precomputed
                         const float* __restrict__ H0U,       // (B,256) row = h0@Uk+bk
                         float* __restrict__ out)             // (B,64)
{
  __shared__ float s_ue[Tsz * Msz];   // swizzled: [t*64 + ((m+t)&63)]
  __shared__ float s_h0u[4 * Psz];
  __shared__ float s_wk[4 * Psz];
  __shared__ float s_s0[Psz];
  __shared__ float s_vd[Msz];
  __shared__ float s_bWd[Msz];
  __shared__ float s_wdense[1 + Msz];
  __shared__ float s_x[Tsz];
  __shared__ float s_hc[2 * Psz];     // [h | c]
  __shared__ float s_dproj[Msz];
  __shared__ float s_beta[Tsz];
  __shared__ float s_ctx[Msz];
  __shared__ float s_red[128];

  const int tid  = threadIdx.x;
  const int lane = tid & 31;
  const int wid  = tid >> 5;
  const int b    = blockIdx.x;

  // ---- one-time loads into LDS -------------------------------------------
  for (int i = tid; i < Tsz * Msz; i += 128) {
    const int tt = i >> 6, m = i & 63;
    s_ue[tt * 64 + ((m + tt) & 63)] = Ue[((size_t)b * Tsz + tt) * Msz + m];
  }
  for (int i = tid; i < 256; i += 128) {
    s_h0u[i] = H0U[(size_t)b * 256 + i];
    s_wk[i]  = Wk[i];
  }
  if (tid < 64) {
    s_s0[tid]  = s0[(size_t)b * Psz + tid];
    s_vd[tid]  = vd[tid];
    s_bWd[tid] = bWd[tid];
    s_ctx[tid] = 0.0f;                       // carry0 ctx = zeros
  }
  if (tid < 65)  s_wdense[tid] = Wdense[tid];
  if (tid < 127) s_x[tid] = data[(size_t)b * 127 + tid];
  const float c_bdense = bdense_p[0];
  const float c_bvd    = bvd_p[0];
  __syncthreads();

  // ---- sequential scan over t --------------------------------------------
  for (int t = 0; t < Tsz - 1; ++t) {
    // Phase 1a: y = x*Wd0 + <ctx, Wd[1:]> + b   (wave-shuffle reduction)
    float prod = (tid < 64) ? s_ctx[tid] * s_wdense[1 + tid] : 0.0f;
    float ws = wave_sum(prod);
    if (lane == 0) s_red[wid] = ws;
    __syncthreads();
    const float y = s_red[0] + s_red[1] + s_x[t] * s_wdense[0] + c_bdense;

    // Phase 1b: LSTM gates (recurrent term precomputed in H0U)
    if (tid < 64) {
      const int p = tid;
      const float zi = fmaf(y, s_wk[p],       s_h0u[p]);
      const float zf = fmaf(y, s_wk[64 + p],  s_h0u[64 + p]);
      const float zg = fmaf(y, s_wk[128 + p], s_h0u[128 + p]);
      const float zo = fmaf(y, s_wk[192 + p], s_h0u[192 + p]);
      const float c  = sigf(zf) * s_s0[p] + sigf(zi) * tanhf(zg);
      const float h  = sigf(zo) * tanhf(c);
      s_hc[p]      = h;
      s_hc[64 + p] = c;
    }
    __syncthreads();

    // Phase 2: dproj = [h|c] @ Wd + bWd   (2 threads per output column)
    {
      const int part = tid >> 6, m = tid & 63;
      const float* wdp = Wd + (size_t)(part * 64) * 64 + m;
      float acc = 0.0f;
      #pragma unroll 4
      for (int k = 0; k < 64; ++k)
        acc = fmaf(s_hc[part * 64 + k], wdp[(size_t)k * 64], acc);
      s_red[tid] = acc;
    }
    __syncthreads();
    if (tid < 64) s_dproj[tid] = s_red[tid] + s_red[64 + tid] + s_bWd[tid];
    __syncthreads();

    // Phase 3: scores + softmax (one thread per t'; swizzled LDS, no bank conflict)
    float e_val = c_bvd;
    {
      const float* uep = s_ue + tid * 64;
      #pragma unroll 4
      for (int m = 0; m < 64; ++m) {
        const float u = uep[(m + tid) & 63];
        e_val = fmaf(tanhf(s_dproj[m] + u), s_vd[m], e_val);
      }
    }
    {
      const float wm = wave_max(e_val);
      if (lane == 0) s_red[4 + wid] = wm;
      __syncthreads();
      const float gmax = fmaxf(fmaxf(s_red[4], s_red[5]), fmaxf(s_red[6], s_red[7]));
      const float ex   = __expf(e_val - gmax);
      const float wsum = wave_sum(ex);
      if (lane == 0) s_red[8 + wid] = wsum;
      __syncthreads();
      const float tot = s_red[8] + s_red[9] + s_red[10] + s_red[11];
      s_beta[tid] = ex / tot;
    }
    __syncthreads();

    // Phase 4: ctx = beta @ enc[b]   (2 threads per column; coalesced L2 reads)
    {
      const int part = tid >> 6, m = tid & 63;
      const float* ep = enc + ((size_t)b * Tsz + part * 64) * Msz + m;
      float acc = 0.0f;
      #pragma unroll 4
      for (int tt = 0; tt < 64; ++tt)
        acc = fmaf(s_beta[part * 64 + tt], ep[(size_t)tt * Msz], acc);
      s_red[tid] = acc;
    }
    __syncthreads();
    if (tid < 64) s_ctx[tid] = s_red[tid] + s_red[64 + tid];
    __syncthreads();
  }

  // ---- final head: out = (([h|ctx]@Wvb)+bvb)*Wb + bWb --------------------
  float v = (tid < 64) ? s_hc[tid] * Wvb[tid] : s_ctx[tid - 64] * Wvb[tid];
  const float ws2 = wave_sum(v);
  if (lane == 0) s_red[wid] = ws2;
  __syncthreads();
  const float sc = s_red[0] + s_red[1] + s_red[2] + s_red[3] + bvb_p[0];
  if (tid < 64) out[(size_t)b * Psz + tid] = fmaf(sc, Wb[tid], bWb[tid]);
}

// ---------------------------------------------------------------------------
extern "C" void kernel_launch(void* const* d_in, const int* in_sizes, int n_in,
                              void* d_out, int out_size, void* d_ws, size_t ws_size,
                              hipStream_t stream) {
  const float* data   = (const float*)d_in[0];
  const float* enc    = (const float*)d_in[1];
  const float* h0     = (const float*)d_in[2];
  const float* s0     = (const float*)d_in[3];
  const float* Wdense = (const float*)d_in[4];
  const float* bdense = (const float*)d_in[5];
  const float* Wk     = (const float*)d_in[6];
  const float* Uk     = (const float*)d_in[7];
  const float* bk     = (const float*)d_in[8];
  const float* Wd     = (const float*)d_in[9];
  const float* bWd    = (const float*)d_in[10];
  const float* Ud     = (const float*)d_in[11];
  const float* bUd    = (const float*)d_in[12];
  const float* vd     = (const float*)d_in[13];
  const float* bvd    = (const float*)d_in[14];
  const float* Wvb    = (const float*)d_in[15];
  const float* bvb    = (const float*)d_in[16];
  const float* Wb     = (const float*)d_in[17];
  const float* bWb    = (const float*)d_in[18];
  float* out = (float*)d_out;

  float* Ue  = (float*)d_ws;                                   // 8 MB
  float* H0U = Ue + (size_t)Bsz * Tsz * Msz;                   // 256 KB

  // Loop-invariant GEMMs on the matrix path (WMMA bf16 -> f32 accumulate)
  // Ue: 2048x4 tiles = 8192 waves -> 1024 blocks of 8 waves
  ue_gemm_kernel<<<1024, 256, 0, stream>>>(enc, Ud, bUd, Ue);
  // H0U: 16x16 tiles = 256 waves -> 32 blocks of 8 waves
  h0u_gemm_kernel<<<32, 256, 0, stream>>>(h0, Uk, bk, H0U);

  // Persistent scan: one workgroup per batch row, 4 waves each
  decoder_scan_kernel<<<Bsz, 128, 0, stream>>>(
      data, enc, s0, Wdense, bdense, Wk, Wd, bWd, vd, bvd,
      Wvb, bvb, Wb, bWb, Ue, H0U, out);
}
